// GNN3_Multisolvent_embedding_54331336294598
// MI455X (gfx1250) — compile-verified, matrix operand order checked by reading, not run on hardware
//
#include <hip/hip_runtime.h>

#define HID 64
#define HTOKD 128
#define OFFSETC 0.0195141f
#define KEC 138.935485f
#define ALPHAC 1.0f
#define BETAC 0.8f
#define GAMMAC 4.85f
#define RADC 0.6f
#define PIF 3.14159265358979f

typedef __attribute__((ext_vector_type(16))) _Float16 v16h;
typedef __attribute__((ext_vector_type(8)))  _Float16 v8h;
typedef __attribute__((ext_vector_type(8)))  float    v8f;

__device__ __forceinline__ float sigf(float x) { return 1.f / (1.f + expf(-x)); }
__device__ __forceinline__ float siluf(float x) { return x * sigf(x); }
__device__ __forceinline__ float dsiluf(float x) { float s = sigf(x); return s * (1.f + x * (1.f - s)); }

__device__ float block_reduce256(float v, float* red) {
    int t = threadIdx.x;
    red[t] = v; __syncthreads();
    for (int s = 128; s > 0; s >>= 1) { if (t < s) red[t] += red[t + s]; __syncthreads(); }
    float r = red[0]; __syncthreads();
    return r;
}

// ---------------- WMMA GEMM: C[M,N] = act(A[M,K] @ Bpacked + bias) ----------------
// A: f16 row-major, lda (halves) multiple of 16. Bpacked: fragment-ordered f16.
// Block = 128 threads = 4 waves; wave w computes rows [blk*64+16w, +16) x all N tiles.
template<int ACT, int WRITE_PRE>
__global__ void gemm_ws(const _Float16* __restrict__ A, int lda,
                        const _Float16* __restrict__ Bp,
                        const float* __restrict__ bias,
                        _Float16* __restrict__ Cout,
                        _Float16* __restrict__ Cpre,
                        int ldc, int M, int Kt, int Nt) {
    const int lane = threadIdx.x & 31;
    const int wave = threadIdx.x >> 5;
    const int row0 = blockIdx.x * 64 + wave * 16;
    const int g = lane >> 4;
    const int rl = lane & 15;
    int arow = row0 + rl;
    if (arow > M - 1) arow = M - 1;            // clamp: keeps EXEC full for WMMA
    const _Float16* arp = A + (size_t)arow * lda;
    for (int nt = 0; nt < Nt; ++nt) {
        v8f acc = {0.f, 0.f, 0.f, 0.f, 0.f, 0.f, 0.f, 0.f};
        for (int kt = 0; kt < Kt; ++kt) {
            const _Float16* ap = arp + kt * 32 + g * 8;
            v8h alo = *(const v8h*)(ap);
            v8h ahi = *(const v8h*)(ap + 16);
            v16h a;
#pragma unroll
            for (int e = 0; e < 8; ++e) { a[e] = alo[e]; a[e + 8] = ahi[e]; }
            v16h b = ((const v16h*)Bp)[(size_t)(kt * Nt + nt) * 32 + lane];
            acc = __builtin_amdgcn_wmma_f32_16x16x32_f16(false, a, false, b, (short)0, acc,
                                                         false, false);
        }
        const int col = nt * 16 + rl;
        const float bs = bias ? bias[col] : 0.f;
#pragma unroll
        for (int rr = 0; rr < 8; ++rr) {
            const int row = row0 + rr + 8 * g;
            if (row < M) {
                float pre = acc[rr] + bs;
                if (WRITE_PRE) Cpre[(size_t)row * ldc + col] = (_Float16)pre;
                float o = (ACT == 1) ? siluf(pre) : pre;
                Cout[(size_t)row * ldc + col] = (_Float16)o;
            }
        }
    }
}

// ---------------- weight packing into WMMA fragment order ----------------
__global__ void pack_wmat(const float* __restrict__ W, int wr, int wc, int transpose,
                          int Kpad, int Npad, _Float16* __restrict__ out) {
    int Nt = Npad >> 4;
    int Kt = Kpad >> 5;
    int total = Kt * Nt * 512;
    int id = blockIdx.x * blockDim.x + threadIdx.x;
    if (id >= total) return;
    int e = id & 15;
    int lane = (id >> 4) & 31;
    int tile = id >> 9;
    int nt = tile % Nt;
    int kt = tile / Nt;
    int k = kt * 32 + (e & 7) + 8 * (lane >> 4) + 16 * (e >> 3);
    int n = nt * 16 + (lane & 15);
    float v = 0.f;
    if (!transpose) { if (k < wr && n < wc) v = W[(size_t)k * wc + n]; }
    else            { if (k < wc && n < wr) v = W[(size_t)n * wc + k]; }
    out[id] = (_Float16)v;
}

__global__ void pack_bias(const float* __restrict__ b, int n, int npad, float* __restrict__ out) {
    int i = blockIdx.x * blockDim.x + threadIdx.x;
    if (i < npad) out[i] = (i < n) ? b[i] : 0.f;
}

// ---------------- small elementwise / gather kernels ----------------
__global__ void atom_prep(const float* __restrict__ af, const float* __restrict__ sdiel,
                          int Nn, float* __restrict__ prefA, float* __restrict__ sA) {
    int i = blockIdx.x * blockDim.x + threadIdx.x;
    if (i >= Nn) return;
    prefA[i] = -0.5f * KEC * (1.f - 1.f / sdiel[i]);
    sA[i] = af[i * 3 + 2] * af[i * 3 + 1];
}

__global__ void build_h0(const float* __restrict__ af, const float* __restrict__ Bb, int Nn,
                         _Float16* __restrict__ h0) {
    int i = blockIdx.x * blockDim.x + threadIdx.x;
    if (i >= Nn) return;
    h0[i * 8 + 0] = (_Float16)Bb[i];
    h0[i * 8 + 1] = (_Float16)af[i * 3 + 0];
    h0[i * 8 + 2] = (_Float16)af[i * 3 + 1];
    for (int c = 3; c < 8; ++c) h0[i * 8 + c] = (_Float16)0.f;
}

__global__ void gather_semb(const float* __restrict__ semb, const int* __restrict__ solv, int Nn,
                            _Float16* __restrict__ Tin) {
    int t = blockIdx.x * blockDim.x + threadIdx.x;
    if (t >= Nn * HID) return;
    int i = t / HID, c = t % HID;
    Tin[t] = (_Float16)semb[(size_t)solv[i] * HID + c];
}

__global__ void gather_edge_x(const int* __restrict__ gsrc, const int* __restrict__ gdst, int E,
                              const _Float16* __restrict__ h, int ldh, int C,
                              _Float16* __restrict__ X, int ldx) {
    size_t tid = (size_t)blockIdx.x * blockDim.x + threadIdx.x;
    size_t total = (size_t)E * ldx;
    if (tid >= total) return;
    int e = (int)(tid / ldx);
    int c = (int)(tid % ldx);
    float v = 0.f;
    if (c < C) v = (float)h[(size_t)gdst[e] * ldh + c];
    else if (c < 2 * C) v = (float)h[(size_t)gsrc[e] * ldh + (c - C)];
    X[tid] = (_Float16)v;
}

__global__ void concat_uin(const float* __restrict__ agg, const _Float16* __restrict__ Tb, int Nn,
                           _Float16* __restrict__ Uin) {
    int t = blockIdx.x * blockDim.x + threadIdx.x;
    if (t >= Nn * 192) return;
    int i = t / 192, c = t % 192;
    Uin[t] = (c < HID) ? (_Float16)agg[(size_t)i * HID + c]
                       : Tb[(size_t)i * HTOKD + (c - HID)];
}

__global__ void ew_dsilu_hh(const _Float16* __restrict__ a, const _Float16* __restrict__ pre,
                            _Float16* __restrict__ o, size_t n) {
    size_t t = (size_t)blockIdx.x * blockDim.x + threadIdx.x;
    if (t >= n) return;
    o[t] = (_Float16)((float)a[t] * dsiluf((float)pre[t]));
}

__global__ void ew_dsilu_fh(const float* __restrict__ a, const _Float16* __restrict__ pre,
                            _Float16* __restrict__ o, size_t n) {
    size_t t = (size_t)blockIdx.x * blockDim.x + threadIdx.x;
    if (t >= n) return;
    o[t] = (_Float16)(a[t] * dsiluf((float)pre[t]));
}

// ---------------- GNN edge helpers ----------------
__global__ void cut_fwd(const float* __restrict__ pos, const int* __restrict__ gsrc,
                        const int* __restrict__ gdst, int E, float* __restrict__ cut) {
    int e = blockIdx.x * blockDim.x + threadIdx.x;
    if (e >= E) return;
    int s = gsrc[e], d = gdst[e];
    float dx = pos[s * 3 + 0] - pos[d * 3 + 0];
    float dy = pos[s * 3 + 1] - pos[d * 3 + 1];
    float dz = pos[s * 3 + 2] - pos[d * 3 + 2];
    float gd = sqrtf(dx * dx + dy * dy + dz * dz);
    float x = fminf(gd, RADC);
    cut[e] = 0.5f * (cosf(PIF * x / RADC) + 1.f);
}

// agg[i,:] = sum over e with dst==i of m2[e,:]*cut[e]  (edges are dst-sorted)
__global__ void agg_kernel(const int* __restrict__ gdst, int E, const _Float16* __restrict__ m2,
                           const float* __restrict__ cut, float* __restrict__ agg) {
    int i = blockIdx.x;
    int c = threadIdx.x;  // 64 threads
    int lo, hi;
    { int a = 0, b = E; while (a < b) { int m = (a + b) >> 1; if (gdst[m] < i) a = m + 1; else b = m; } lo = a; }
    { int a = lo, b = E; while (a < b) { int m = (a + b) >> 1; if (gdst[m] <= i) a = m + 1; else b = m; } hi = a; }
    float s = 0.f;
    for (int e = lo; e < hi; ++e) s += (float)m2[(size_t)e * HID + c] * cut[e];
    agg[(size_t)i * HID + c] = s;
}

// g_cut[e] += dot(gAgg[dst],m2[e]); overwrite m2 with gPreM2 = gAgg[dst]*cut*silu'(pm2)
__global__ void edge_back1(const int* __restrict__ gdst, int E, const _Float16* __restrict__ gAgg,
                           _Float16* __restrict__ m2, const _Float16* __restrict__ pm2,
                           const float* __restrict__ cut, float* __restrict__ g_cut) {
    int e = blockIdx.x;
    if (e >= E) return;
    int c = threadIdx.x;  // 64
    __shared__ float red[64];
    int node = gdst[e];
    float gm = (float)gAgg[(size_t)node * HID + c];
    float m2v = (float)m2[(size_t)e * HID + c];
    red[c] = gm * m2v;
    __syncthreads();
    for (int s = 32; s > 0; s >>= 1) { if (c < s) red[c] += red[c + s]; __syncthreads(); }
    if (c == 0) g_cut[e] += red[0];
    float gp = gm * cut[e] * dsiluf((float)pm2[(size_t)e * HID + c]);
    m2[(size_t)e * HID + c] = (_Float16)gp;
}

__global__ void scatter_gx(const int* __restrict__ gsrc, const int* __restrict__ gdst, int E,
                           const _Float16* __restrict__ gX, int ldx, int C,
                           float* __restrict__ gH, int ldg) {
    size_t tid = (size_t)blockIdx.x * blockDim.x + threadIdx.x;
    size_t total = (size_t)E * 2 * C;
    if (tid >= total) return;
    int e = (int)(tid / (2 * C));
    int c = (int)(tid % (2 * C));
    float v = (float)gX[(size_t)e * ldx + c];
    if (c < C) atomicAdd(&gH[(size_t)gdst[e] * ldg + c], v);
    else       atomicAdd(&gH[(size_t)gsrc[e] * ldg + (c - C)], v);
}

__global__ void cut_bwd(const float* __restrict__ pos, const int* __restrict__ gsrc,
                        const int* __restrict__ gdst, int E, const float* __restrict__ g_cut,
                        float* __restrict__ gpos) {
    int e = blockIdx.x * blockDim.x + threadIdx.x;
    if (e >= E) return;
    int s = gsrc[e], d = gdst[e];
    float dx = pos[s * 3 + 0] - pos[d * 3 + 0];
    float dy = pos[s * 3 + 1] - pos[d * 3 + 1];
    float dz = pos[s * 3 + 2] - pos[d * 3 + 2];
    float gd = sqrtf(dx * dx + dy * dy + dz * dz);
    if (gd >= RADC || gd <= 0.f) return;
    float dc = -0.5f * (PIF / RADC) * sinf(PIF * gd / RADC);
    float w = g_cut[e] * dc / gd;
    atomicAdd(&gpos[d * 3 + 0], -w * dx);
    atomicAdd(&gpos[d * 3 + 1], -w * dy);
    atomicAdd(&gpos[d * 3 + 2], -w * dz);
    atomicAdd(&gpos[s * 3 + 0], w * dx);
    atomicAdd(&gpos[s * 3 + 1], w * dy);
    atomicAdd(&gpos[s * 3 + 2], w * dz);
}

// ---------------- O(N^2) GB kernels ----------------
__device__ __forceinline__ float Ie_val(float r, float sj, float oi) {
    float U = r + sj;
    if (U <= oi) return 0.f;
    float L = fmaxf(fabsf(r - sj), oi);
    float invL = 1.f / L, invU = 1.f / U;
    return 0.5f * (invL - invU + 0.25f * (r - sj * sj / r) * (invU * invU - invL * invL) +
                   0.5f * logf(L * invU) / r);
}

__device__ __forceinline__ float Ie_dr(float r, float sj, float oi) {
    float U = r + sj;
    if (U <= oi) return 0.f;
    float d = r - sj;
    float ad = fabsf(d);
    float L, Lp;
    if (ad > oi) { L = ad; Lp = (d > 0.f) ? 1.f : -1.f; } else { L = oi; Lp = 0.f; }
    float invL = 1.f / L, invU = 1.f / U;
    float invL2 = invL * invL, invU2 = invU * invU;
    float t1 = -Lp * invL2 + invU2;
    float a = r - sj * sj / r;
    float da = 1.f + sj * sj / (r * r);
    float t2 = 0.25f * (da * (invU2 - invL2) + a * (-2.f * invU2 * invU + 2.f * Lp * invL2 * invL));
    float lg = logf(L * invU);
    float t3 = 0.5f * ((Lp * invL - invU) / r - lg / (r * r));
    return 0.5f * (t1 + t2 + t3);
}

__global__ void born_I(const float* __restrict__ pos, const float* __restrict__ af,
                       const float* __restrict__ sA, int Nn, float* __restrict__ I,
                       float* __restrict__ psi, float* __restrict__ Bb) {
    __shared__ float red[256];
    int i = blockIdx.x;
    float px = pos[i * 3 + 0], py = pos[i * 3 + 1], pz = pos[i * 3 + 2];
    float oi = af[i * 3 + 1];
    float acc = 0.f;
    for (int j = threadIdx.x; j < Nn; j += 256) {
        if (j == i) continue;
        float dx = pos[j * 3 + 0] - px, dy = pos[j * 3 + 1] - py, dz = pos[j * 3 + 2] - pz;
        float r = sqrtf(dx * dx + dy * dy + dz * dz);
        acc += Ie_val(r, sA[j], oi);
    }
    float tot = block_reduce256(acc, red);
    if (threadIdx.x == 0) {
        I[i] = tot;
        float ps = tot * oi;
        psi[i] = ps;
        float rho = oi + OFFSETC;
        float th = tanhf(ALPHAC * ps - BETAC * ps * ps + GAMMAC * ps * ps * ps);
        Bb[i] = 1.f / (1.f / oi - th / rho);
    }
}

// pair energy + dE/dBsc (A) + r-path force contribution (folds both ordered directions)
__global__ void pair_energy_back(const float* __restrict__ pos, const float* __restrict__ af,
                                 const float* __restrict__ prefA, const float* __restrict__ Bsc,
                                 int Nn, float* __restrict__ Eout, float* __restrict__ A,
                                 float* __restrict__ gpos) {
    __shared__ float red[256];
    int i = blockIdx.x;
    float px = pos[i * 3 + 0], py = pos[i * 3 + 1], pz = pos[i * 3 + 2];
    float qi = af[i * 3 + 0], Bi = Bsc[i], pfi = prefA[i];
    float accE = 0.f, accA = 0.f, fx = 0.f, fy = 0.f, fz = 0.f;
    for (int j = threadIdx.x; j < Nn; j += 256) {
        if (j == i) continue;
        float dx = px - pos[j * 3 + 0], dy = py - pos[j * 3 + 1], dz = pz - pos[j * 3 + 2];
        float r2 = dx * dx + dy * dy + dz * dz;
        float r = sqrtf(r2);
        float qj = af[j * 3 + 0], Bj = Bsc[j], pfj = prefA[j];
        float x = Bi * Bj;
        float w = expf(-r2 / (4.f * x));
        float f = sqrtf(r2 + x * w);
        accE += pfi * qi * qj / f;                 // only dst==i direction counted for energy
        float P = (pfi + pfj) * qi * qj;           // both directions share r, x, f
        float gef = -P / (f * f);
        float dfdx = w * (1.f + r2 / (4.f * x)) / (2.f * f);
        float dfdr = r * (1.f - 0.25f * w) / f;
        accA += gef * dfdx * Bj;
        float fr = gef * dfdr / r;
        fx += fr * dx; fy += fr * dy; fz += fr * dz;
    }
    float tE = block_reduce256(accE, red);
    float tA = block_reduce256(accA, red);
    float tx = block_reduce256(fx, red);
    float ty = block_reduce256(fy, red);
    float tz = block_reduce256(fz, red);
    if (threadIdx.x == 0) {
        atomicAdd(Eout, tE);
        A[i] = tA;
        gpos[i * 3 + 0] += tx;
        gpos[i * 3 + 1] += ty;
        gpos[i * 3 + 2] += tz;
    }
}

// dE/dI -> force (both ordered directions evaluated per unordered pair, block-local)
__global__ void born_back(const float* __restrict__ pos, const float* __restrict__ af,
                          const float* __restrict__ sA, const float* __restrict__ D, int Nn,
                          float* __restrict__ gpos) {
    __shared__ float red[256];
    int i = blockIdx.x;
    float px = pos[i * 3 + 0], py = pos[i * 3 + 1], pz = pos[i * 3 + 2];
    float oi = af[i * 3 + 1], si = sA[i], Di = D[i];
    float fx = 0.f, fy = 0.f, fz = 0.f;
    for (int j = threadIdx.x; j < Nn; j += 256) {
        if (j == i) continue;
        float dx = px - pos[j * 3 + 0], dy = py - pos[j * 3 + 1], dz = pz - pos[j * 3 + 2];
        float r = sqrtf(dx * dx + dy * dy + dz * dz);
        float gij = Ie_dr(r, sA[j], oi) * Di;            // I_i <- j
        float gji = Ie_dr(r, si, af[j * 3 + 1]) * D[j];  // I_j <- i
        float fr = (gij + gji) / r;
        fx += fr * dx; fy += fr * dy; fz += fr * dz;
    }
    float tx = block_reduce256(fx, red);
    float ty = block_reduce256(fy, red);
    float tz = block_reduce256(fz, red);
    if (threadIdx.x == 0) {
        gpos[i * 3 + 0] += tx;
        gpos[i * 3 + 1] += ty;
        gpos[i * 3 + 2] += tz;
    }
}

// ---------------- per-node energy epilogue & gradient seeds ----------------
__global__ void compute_bsc(const float* __restrict__ Bb, const _Float16* __restrict__ outF,
                            float* __restrict__ Bsc, int Nn) {
    int i = blockIdx.x * blockDim.x + threadIdx.x;
    if (i >= Nn) return;
    float c = (float)outF[i * 16 + 0];
    Bsc[i] = Bb[i] * (0.1f + 1.8f * sigf(c));
}

__global__ void node_post(const float* __restrict__ af, const float* __restrict__ prefA,
                          const float* __restrict__ gemb, const int* __restrict__ solv,
                          const float* __restrict__ Bb, const float* __restrict__ Bsc,
                          const _Float16* __restrict__ outF, float* __restrict__ A,
                          float* __restrict__ Eacc, _Float16* __restrict__ gO2, int Nn) {
    int i = blockIdx.x * blockDim.x + threadIdx.x;
    if (i >= Nn) return;
    float q = af[i * 3 + 0], o = af[i * 3 + 1], rho = o + OFFSETC;
    float pref = prefA[i];
    float c = (float)outF[i * 16 + 0], s = (float)outF[i * 16 + 1];
    float sc = sigf(c), ss = sigf(s);
    float gam = gemb[solv[i]];
    float rr = (rho + 0.14f) * (rho + 0.14f);
    atomicAdd(Eacc, pref * q * q / Bsc[i] + gam * ss * rr);
    float At = A[i] - pref * q * q / (Bsc[i] * Bsc[i]);   // add self-energy dE/dBsc
    A[i] = At;
    float gc = At * Bb[i] * 1.8f * sc * (1.f - sc);
    float gs = gam * ss * (1.f - ss) * rr;
    gO2[i * 32 + 0] = (_Float16)gc;
    gO2[i * 32 + 1] = (_Float16)gs;
    for (int k = 2; k < 32; ++k) gO2[i * 32 + k] = (_Float16)0.f;
}

__global__ void compute_D(const float* __restrict__ A, const float* __restrict__ gH0,
                          const _Float16* __restrict__ outF, const float* __restrict__ Bb,
                          const float* __restrict__ psi, const float* __restrict__ af,
                          float* __restrict__ D, int Nn) {
    int i = blockIdx.x * blockDim.x + threadIdx.x;
    if (i >= Nn) return;
    float c = (float)outF[i * 16 + 0];
    float gB = gH0[i * 16 + 0] + A[i] * (0.1f + 1.8f * sigf(c));
    float o = af[i * 3 + 1], rho = o + OFFSETC, ps = psi[i];
    float th = tanhf(ALPHAC * ps - BETAC * ps * ps + GAMMAC * ps * ps * ps);
    float dBdth = Bb[i] * Bb[i] / rho;
    float dthdps = (1.f - th * th) * (ALPHAC - 2.f * BETAC * ps + 3.f * GAMMAC * ps * ps);
    D[i] = gB * dBdth * dthdps * o;
}

__global__ void finalize_forces(const float* __restrict__ gpos, float* __restrict__ out, int Nn) {
    int t = blockIdx.x * blockDim.x + threadIdx.x;
    if (t < Nn * 3) out[1 + t] = -gpos[t];
}

// ============================= host =============================
extern "C" void kernel_launch(void* const* d_in, const int* in_sizes, int n_in,
                              void* d_out, int out_size, void* d_ws, size_t ws_size,
                              hipStream_t stream) {
    const float* pos   = (const float*)d_in[0];
    const float* af    = (const float*)d_in[1];
    const float* sdiel = (const float*)d_in[2];
    const float* semb  = (const float*)d_in[3];
    const float* gemb  = (const float*)d_in[4];
    const int* solv    = (const int*)d_in[n_in - 3];
    const int* gnn     = (const int*)d_in[n_in - 1];
    const int N = in_sizes[0] / 3;
    const int E = in_sizes[n_in - 1] / 2;
    const int* gsrc = gnn;
    const int* gdst = gnn + E;
    const int Epad = ((E + 63) / 64) * 64;
    float* Eout = (float*)d_out;

    // ---- resolve layer weight pointers (order: Wm1,bm1,Wm2,bm2,Wt,bt,Wu1,bu1,Wu2,bu2)
    const int inch[3] = {6, 128, 128};
    const int outch[3] = {64, 64, 2};
    const float* W[3][10];
    if (n_in >= 38) {
        bool alpha = (in_sizes[6] != 64);  // alphabetical leaf order vs insertion order
        for (int L = 0; L < 3; ++L) {
            int b = 5 + L * 10;
            if (!alpha) {
                for (int k = 0; k < 10; ++k) W[L][k] = (const float*)d_in[b + k];
            } else {  // Wm1,Wm2,Wt,Wu1,Wu2,bm1,bm2,bt,bu1,bu2
                W[L][0] = (const float*)d_in[b + 0]; W[L][1] = (const float*)d_in[b + 5];
                W[L][2] = (const float*)d_in[b + 1]; W[L][3] = (const float*)d_in[b + 6];
                W[L][4] = (const float*)d_in[b + 2]; W[L][5] = (const float*)d_in[b + 7];
                W[L][6] = (const float*)d_in[b + 3]; W[L][7] = (const float*)d_in[b + 8];
                W[L][8] = (const float*)d_in[b + 4]; W[L][9] = (const float*)d_in[b + 9];
            }
        }
    } else {  // single concatenated blob, insertion order
        const float* blob = (const float*)d_in[5];
        size_t o = 0;
        for (int L = 0; L < 3; ++L) {
            W[L][0] = blob + o; o += (size_t)inch[L] * 64;
            W[L][1] = blob + o; o += 64;
            W[L][2] = blob + o; o += 64 * 64;
            W[L][3] = blob + o; o += 64;
            W[L][4] = blob + o; o += 64 * 128;
            W[L][5] = blob + o; o += 128;
            W[L][6] = blob + o; o += 192 * 64;
            W[L][7] = blob + o; o += 64;
            W[L][8] = blob + o; o += (size_t)64 * outch[L];
            W[L][9] = blob + o; o += outch[L];
        }
    }

    // ---- workspace bump allocator
    char* wsp = (char*)d_ws;
    size_t off = 0;
    auto alloc = [&](size_t bytes) -> void* {
        void* p = wsp + off;
        off = (off + bytes + 255) & ~(size_t)255;
        return p;
    };
    auto fbuf = [&](size_t n) { return (float*)alloc(n * 4); };
    auto hbuf = [&](size_t n) { return (_Float16*)alloc(n * 2); };

    float *I = fbuf(N), *psi = fbuf(N), *Bb = fbuf(N), *Bsc = fbuf(N);
    float *A = fbuf(N), *D = fbuf(N), *prefA = fbuf(N), *sA = fbuf(N);
    float *cut = fbuf(Epad), *g_cut = fbuf(Epad);
    float *agg = fbuf((size_t)N * HID);
    float *gH2 = fbuf((size_t)N * HID), *gH1 = fbuf((size_t)N * HID), *gH0 = fbuf((size_t)N * 16);
    float *gpos = fbuf((size_t)N * 3);

    _Float16 *h0 = hbuf((size_t)N * 8);
    _Float16 *h1 = hbuf((size_t)N * HID), *h2 = hbuf((size_t)N * HID);
    _Float16 *Tin = hbuf((size_t)N * HID), *Tbuf = hbuf((size_t)N * HTOKD);
    _Float16 *Uin = hbuf((size_t)N * 192), *Ubuf = hbuf((size_t)N * HID);
    _Float16 *preU[3] = {hbuf((size_t)N * HID), hbuf((size_t)N * HID), hbuf((size_t)N * HID)};
    _Float16 *preOut[2] = {hbuf((size_t)N * HID), hbuf((size_t)N * HID)};
    _Float16 *outF = hbuf((size_t)N * 16);
    _Float16 *gOx = hbuf((size_t)N * HID), *gO2 = hbuf((size_t)N * 32);
    _Float16 *gU = hbuf((size_t)N * HID), *gPreU = hbuf((size_t)N * HID);
    _Float16 *gAggB = hbuf((size_t)N * HID);
    _Float16 *X = hbuf((size_t)Epad * 128);
    _Float16 *m1 = hbuf((size_t)Epad * HID), *pm1 = hbuf((size_t)Epad * HID);
    _Float16 *m2 = hbuf((size_t)Epad * HID), *pm2 = hbuf((size_t)Epad * HID);

    const int Kx[3] = {32, 128, 128};       // padded X widths (K of m1 gemm)
    const int NoutPad[3] = {64, 64, 16};    // padded Wu2 N
    const int KWu2T[3] = {64, 64, 32};      // padded K for Wu2^T gemm
    const int Nm1T[3] = {16, 128, 128};     // padded N for Wm1^T gemm
    _Float16 *pWm1[3], *pWm2[3], *pWt[3], *pWu1[3], *pWu2[3];
    _Float16 *pWm1T[3], *pWm2T[3], *pWu2T[3], *pWu1Tt[3];
    float *bm1p[3], *bm2p[3], *btp[3], *bu1p[3], *bu2p[3];
    for (int L = 0; L < 3; ++L) {
        pWm1[L] = hbuf((size_t)Kx[L] * 64);
        pWm2[L] = hbuf(64 * 64);
        pWt[L] = hbuf(64 * 128);
        pWu1[L] = hbuf(192 * 64);
        pWu2[L] = hbuf((size_t)64 * NoutPad[L]);
        pWm1T[L] = hbuf((size_t)64 * Nm1T[L]);
        pWm2T[L] = hbuf(64 * 64);
        pWu2T[L] = hbuf((size_t)KWu2T[L] * 64);
        pWu1Tt[L] = hbuf(64 * 64);
        bm1p[L] = fbuf(64); bm2p[L] = fbuf(64); btp[L] = fbuf(128);
        bu1p[L] = fbuf(64); bu2p[L] = fbuf(NoutPad[L]);
    }

    // ---- zero init
    (void)hipMemsetAsync(Eout, 0, sizeof(float), stream);
    (void)hipMemsetAsync(gpos, 0, (size_t)N * 3 * 4, stream);
    (void)hipMemsetAsync(gH2, 0, (size_t)N * HID * 4, stream);
    (void)hipMemsetAsync(gH1, 0, (size_t)N * HID * 4, stream);
    (void)hipMemsetAsync(gH0, 0, (size_t)N * 16 * 4, stream);
    (void)hipMemsetAsync(g_cut, 0, (size_t)Epad * 4, stream);

    // ---- pack weights
    auto pack = [&](const float* w, int wr, int wc, int tr, int Kpad, int Npad, _Float16* dst) {
        int total = (Kpad / 32) * (Npad / 16) * 512;
        pack_wmat<<<(total + 255) / 256, 256, 0, stream>>>(w, wr, wc, tr, Kpad, Npad, dst);
    };
    auto packb = [&](const float* b, int n, int npad, float* dst) {
        pack_bias<<<(npad + 255) / 256, 256, 0, stream>>>(b, n, npad, dst);
    };
    for (int L = 0; L < 3; ++L) {
        pack(W[L][0], inch[L], 64, 0, Kx[L], 64, pWm1[L]);
        pack(W[L][2], 64, 64, 0, 64, 64, pWm2[L]);
        pack(W[L][4], 64, 128, 0, 64, 128, pWt[L]);
        pack(W[L][6], 192, 64, 0, 192, 64, pWu1[L]);
        pack(W[L][8], 64, outch[L], 0, 64, NoutPad[L], pWu2[L]);
        pack(W[L][0], inch[L], 64, 1, 64, Nm1T[L], pWm1T[L]);
        pack(W[L][2], 64, 64, 1, 64, 64, pWm2T[L]);
        pack(W[L][8], 64, outch[L], 1, KWu2T[L], 64, pWu2T[L]);
        pack(W[L][6], 192, 64, 1, 64, 64, pWu1Tt[L]);  // (Wu1[:64,:])^T via N-truncation
        packb(W[L][1], 64, 64, bm1p[L]);
        packb(W[L][3], 64, 64, bm2p[L]);
        packb(W[L][5], 128, 128, btp[L]);
        packb(W[L][7], 64, 64, bu1p[L]);
        packb(W[L][9], outch[L], NoutPad[L], bu2p[L]);
    }

    auto gemm = [&](const _Float16* Am, int lda, const _Float16* Bp, const float* bias,
                    _Float16* Co, _Float16* Cp, int ldc, int M, int K, int Nn, int act) {
        dim3 grid((M + 63) / 64), block(128);
        int Kt = K / 32, Nt = Nn / 16;
        if (act) {
            if (Cp) gemm_ws<1, 1><<<grid, block, 0, stream>>>(Am, lda, Bp, bias, Co, Cp, ldc, M, Kt, Nt);
            else    gemm_ws<1, 0><<<grid, block, 0, stream>>>(Am, lda, Bp, bias, Co, Co, ldc, M, Kt, Nt);
        } else {
            if (Cp) gemm_ws<0, 1><<<grid, block, 0, stream>>>(Am, lda, Bp, bias, Co, Cp, ldc, M, Kt, Nt);
            else    gemm_ws<0, 0><<<grid, block, 0, stream>>>(Am, lda, Bp, bias, Co, Co, ldc, M, Kt, Nt);
        }
    };

    const int TB = 256;
    auto g1 = [&](int n) { return dim3((n + TB - 1) / TB); };

    // ---- forward: Born radii, features, cut
    atom_prep<<<g1(N), TB, 0, stream>>>(af, sdiel, N, prefA, sA);
    born_I<<<N, 256, 0, stream>>>(pos, af, sA, N, I, psi, Bb);
    build_h0<<<g1(N), TB, 0, stream>>>(af, Bb, N, h0);
    if (E > 0) cut_fwd<<<g1(E), TB, 0, stream>>>(pos, gsrc, gdst, E, cut);
    gather_semb<<<g1(N * HID), TB, 0, stream>>>(semb, solv, N, Tin);

    _Float16* hs[3] = {h0, h1, h2};
    const int hld[3] = {8, HID, HID};
    const int Cd[3] = {3, HID, HID};

    // ---- GNN forward
    for (int L = 0; L < 3; ++L) {
        if (E > 0) {
            int nbX = (int)(((size_t)E * Kx[L] + TB - 1) / TB);
            gather_edge_x<<<nbX, TB, 0, stream>>>(gsrc, gdst, E, hs[L], hld[L], Cd[L], X, Kx[L]);
            gemm(X, Kx[L], pWm1[L], bm1p[L], m1, pm1, HID, E, Kx[L], 64, 1);
            gemm(m1, HID, pWm2[L], bm2p[L], m2, pm2, HID, E, 64, 64, 1);
        }
        agg_kernel<<<N, 64, 0, stream>>>(gdst, E, m2, cut, agg);
        gemm(Tin, HID, pWt[L], btp[L], Tbuf, nullptr, HTOKD, N, 64, 128, 1);
        concat_uin<<<g1(N * 192), TB, 0, stream>>>(agg, Tbuf, N, Uin);
        gemm(Uin, 192, pWu1[L], bu1p[L], Ubuf, preU[L], HID, N, 192, 64, 1);
        if (L < 2) gemm(Ubuf, HID, pWu2[L], bu2p[L], hs[L + 1], preOut[L], HID, N, 64, 64, 1);
        else       gemm(Ubuf, HID, pWu2[L], bu2p[L], outF, nullptr, 16, N, 64, 16, 0);
    }

    // ---- energies + dE/dBsc + pair-r forces
    compute_bsc<<<g1(N), TB, 0, stream>>>(Bb, outF, Bsc, N);
    pair_energy_back<<<N, 256, 0, stream>>>(pos, af, prefA, Bsc, N, Eout, A, gpos);
    node_post<<<g1(N), TB, 0, stream>>>(af, prefA, gemb, solv, Bb, Bsc, outF, A, Eout, gO2, N);

    // ---- GNN backward (layers 2 -> 0)
    float* gHl[3] = {gH0, gH1, gH2};
    const int gld[3] = {16, HID, HID};
    for (int L = 2; L >= 0; --L) {
        const _Float16* gO = (L == 2) ? gO2 : gOx;
        const int gOld = (L == 2) ? 32 : HID;
        gemm(gO, gOld, pWu2T[L], nullptr, gU, nullptr, HID, N, KWu2T[L], 64, 0);
        ew_dsilu_hh<<<g1(N * HID), TB, 0, stream>>>(gU, preU[L], gPreU, (size_t)N * HID);
        gemm(gPreU, HID, pWu1Tt[L], nullptr, gAggB, nullptr, HID, N, 64, 64, 0);
        if (E > 0) {
            // recompute message activations
            int nbX = (int)(((size_t)E * Kx[L] + TB - 1) / TB);
            gather_edge_x<<<nbX, TB, 0, stream>>>(gsrc, gdst, E, hs[L], hld[L], Cd[L], X, Kx[L]);
            gemm(X, Kx[L], pWm1[L], bm1p[L], m1, pm1, HID, E, Kx[L], 64, 1);
            gemm(m1, HID, pWm2[L], bm2p[L], m2, pm2, HID, E, 64, 64, 1);
            edge_back1<<<E, 64, 0, stream>>>(gdst, E, gAggB, m2, pm2, cut, g_cut);
            gemm(m2, HID, pWm2T[L], nullptr, pm2, nullptr, HID, E, 64, 64, 0);      // gM1 -> pm2
            int nbE = (int)(((size_t)E * HID + TB - 1) / TB);
            ew_dsilu_hh<<<nbE, TB, 0, stream>>>(pm2, pm1, m1, (size_t)E * HID);     // gPreM1 -> m1
            gemm(m1, HID, pWm1T[L], nullptr, X, nullptr, Kx[L], E, 64, Nm1T[L], 0); // gX -> X
            int nbS = (int)(((size_t)E * 2 * Cd[L] + TB - 1) / TB);
            scatter_gx<<<nbS, TB, 0, stream>>>(gsrc, gdst, E, X, Kx[L], Cd[L], gHl[L], gld[L]);
        }
        if (L > 0)
            ew_dsilu_fh<<<g1(N * HID), TB, 0, stream>>>(gHl[L], preOut[L - 1], gOx, (size_t)N * HID);
    }

    // ---- remaining pos-gradient paths
    compute_D<<<g1(N), TB, 0, stream>>>(A, gH0, outF, Bb, psi, af, D, N);
    born_back<<<N, 256, 0, stream>>>(pos, af, sA, D, N, gpos);
    if (E > 0) cut_bwd<<<g1(E), TB, 0, stream>>>(pos, gsrc, gdst, E, g_cut, gpos);
    finalize_forces<<<g1(N * 3), TB, 0, stream>>>(gpos, Eout, N);
}
